// MultiHeadAttentionBlock_65704409694609
// MI455X (gfx1250) — matrix operand order repaired and checked by reading, not compiled, not run on hardware
//
#include <hip/hip_runtime.h>
#include <hip/hip_bf16.h>

typedef _Float16 half_t;
typedef __attribute__((ext_vector_type(16))) _Float16 v16h;
typedef __attribute__((ext_vector_type(8)))  _Float16 v8h;
typedef __attribute__((ext_vector_type(8)))  float    v8f;

// GCC-style vector matching the async builtin's parameter type exactly:
// 'int __attribute__((__vector_size__(4 * sizeof(int))))'
typedef int gvec4i __attribute__((vector_size(16)));
#define AS_GLOBAL __attribute__((address_space(1)))
#define AS_LDS    __attribute__((address_space(3)))

#define B_ 2
#define S_ 4096
#define D_ 512
#define H_ 8
#define DK_ 64
#define M_ (B_ * S_)   // 8192 rows

static __device__ __forceinline__ v8f wmma32(v16h a, v16h b, v8f c) {
    return __builtin_amdgcn_wmma_f32_16x16x32_f16(false, a, false, b, (short)0, c, false, false);
}

union V16U { v16h v; uint4 q[2]; };
union V8U  { v8h  v; uint4 q;    };

#if __has_builtin(__builtin_amdgcn_global_load_async_to_lds_b128)
#define ASYNC_LDS 1
#endif

static __device__ __forceinline__ void wait_asynccnt0() {
#if defined(ASYNC_LDS)
#if __has_builtin(__builtin_amdgcn_s_wait_asynccnt)
    __builtin_amdgcn_s_wait_asynccnt(0);
#else
    asm volatile("s_wait_asynccnt 0" ::: "memory");
#endif
#endif
}

// ---------------------------------------------------------------------------
// GEMM: Y[m,n] = sum_k X[m,k] * W[n,k] + bias[n]   (X fp32, W fp32 -> Y f16)
// One wave computes a 64x64 macro-tile: 4 A-frags x 4 B-frags = 16 WMMAs per
// 32-wide K step -> high WMMA:VALU density. 128 m-macros x 8 n-macros = 1024
// waves -> 128 blocks of 8 waves.
// ---------------------------------------------------------------------------
__global__ __launch_bounds__(256) void proj_f32h(const float* __restrict__ X,
                                                 const float* __restrict__ W,
                                                 const float* __restrict__ bias,
                                                 half_t* __restrict__ Y) {
    const int lane = threadIdx.x & 31;
    const int wave = threadIdx.x >> 5;
    const int idx  = blockIdx.x * 8 + wave;
    const int nt = idx & 7;
    const int mt = idx >> 3;
    const int m0 = mt * 64, n0 = nt * 64;
    const int g = lane >> 4, c = lane & 15;

    const float* __restrict__ arow[4];
    const float* __restrict__ brow[4];
#pragma unroll
    for (int s = 0; s < 4; s++) {
        arow[s] = X + (size_t)(m0 + 16 * s + c) * D_;
        brow[s] = W + (size_t)(n0 + 16 * s + c) * D_;
    }

    v8f acc[4][4];
#pragma unroll
    for (int s = 0; s < 4; s++)
#pragma unroll
        for (int t = 0; t < 4; t++) acc[s][t] = v8f{};

    for (int k0 = 0; k0 < D_; k0 += 32) {
        v16h A[4];
#pragma unroll
        for (int s = 0; s < 4; s++)
#pragma unroll
            for (int i = 0; i < 8; i++) {
                A[s][i]     = (half_t)arow[s][k0 + 8 * g + i];       // halves 0..7 : K = 8g+i
                A[s][i + 8] = (half_t)arow[s][k0 + 16 + 8 * g + i];  // halves 8..15: K = 16+8g+i
            }
#pragma unroll
        for (int bs = 0; bs < 4; bs++) {
            v16h Bf;
#pragma unroll
            for (int i = 0; i < 16; i++)
                Bf[i] = (half_t)brow[bs][k0 + 8 * g + i + (i >= 8 ? 8 : 0)];  // B halves: K = 8g+i / 16+8g+(i-8)
#pragma unroll
            for (int s = 0; s < 4; s++) acc[s][bs] = wmma32(A[s], Bf, acc[s][bs]);
        }
    }
#pragma unroll
    for (int bs = 0; bs < 4; bs++) {
        const float bv = bias[n0 + 16 * bs + c];
#pragma unroll
        for (int s = 0; s < 4; s++)
#pragma unroll
            for (int r = 0; r < 8; r++)
                Y[(size_t)(m0 + 16 * s + r + 8 * g) * D_ + n0 + 16 * bs + c] =
                    (half_t)(acc[s][bs][r] + bv);
    }
}

// ---------------------------------------------------------------------------
// GEMM: out[m,n] = sum_k Xh[m,k]*W[n,k] + bias[n]  (Xh f16, W fp32 -> out f32)
// Same 64x64 macro-tile structure.
// ---------------------------------------------------------------------------
__global__ __launch_bounds__(256) void proj_h32(const half_t* __restrict__ Xh,
                                                const float* __restrict__ W,
                                                const float* __restrict__ bias,
                                                float* __restrict__ Y) {
    const int lane = threadIdx.x & 31;
    const int wave = threadIdx.x >> 5;
    const int idx  = blockIdx.x * 8 + wave;
    const int nt = idx & 7;
    const int mt = idx >> 3;
    const int m0 = mt * 64, n0 = nt * 64;
    const int g = lane >> 4, c = lane & 15;

    const half_t* __restrict__ arow[4];
    const float*  __restrict__ brow[4];
#pragma unroll
    for (int s = 0; s < 4; s++) {
        arow[s] = Xh + (size_t)(m0 + 16 * s + c) * D_;
        brow[s] = W  + (size_t)(n0 + 16 * s + c) * D_;
    }

    v8f acc[4][4];
#pragma unroll
    for (int s = 0; s < 4; s++)
#pragma unroll
        for (int t = 0; t < 4; t++) acc[s][t] = v8f{};

    for (int k0 = 0; k0 < D_; k0 += 32) {
        v16h A[4];
#pragma unroll
        for (int s = 0; s < 4; s++) {
            V16U a;
            a.q[0] = *reinterpret_cast<const uint4*>(arow[s] + k0 + 8 * g);
            a.q[1] = *reinterpret_cast<const uint4*>(arow[s] + k0 + 16 + 8 * g);
            A[s] = a.v;
        }
#pragma unroll
        for (int bs = 0; bs < 4; bs++) {
            v16h Bf;
#pragma unroll
            for (int i = 0; i < 16; i++)
                Bf[i] = (half_t)brow[bs][k0 + 8 * g + i + (i >= 8 ? 8 : 0)];
#pragma unroll
            for (int s = 0; s < 4; s++) acc[s][bs] = wmma32(A[s], Bf, acc[s][bs]);
        }
    }
#pragma unroll
    for (int bs = 0; bs < 4; bs++) {
        const float bv = bias[n0 + 16 * bs + c];
#pragma unroll
        for (int s = 0; s < 4; s++)
#pragma unroll
            for (int r = 0; r < 8; r++)
                Y[(size_t)(m0 + 16 * s + r + 8 * g) * D_ + n0 + 16 * bs + c] =
                    acc[s][bs][r] + bv;
    }
}

// ---------------------------------------------------------------------------
// Flash attention (faithful to reference: masked scores = -1e-9, NOT -inf, so
// every query attends to all 4096 keys). One wave = 16 q-rows; one block =
// 8 waves = 128 q-rows of one (b,h). K tile staged to LDS with async-to-LDS
// b128 (ASYNCcnt) when available; V tile transposed into LDS; online softmax
// in fp32; P re-layout via per-wave LDS scratch.
// ---------------------------------------------------------------------------
#define KP 72   // Ksh pitch in halves (64 + 8)
#define VP 40   // VshT pitch in halves (32 + 8)
#define PP 40   // Psh pitch in halves (32 + 8)

__global__ __launch_bounds__(256) void attn_kernel(const half_t* __restrict__ Qp,
                                                   const half_t* __restrict__ Kp,
                                                   const half_t* __restrict__ Vp,
                                                   half_t* __restrict__ Xo) {
    __shared__ half_t Ksh[32 * KP];        // [j][d]
    __shared__ half_t Vsh[64 * VP];        // [d][j]  (transposed at staging)
    __shared__ half_t Psh[8 * 16 * PP];    // per-wave [row][j]

    const int lane = threadIdx.x & 31;
    const int wave = threadIdx.x >> 5;
    const int blk  = blockIdx.x;           // b*256 + h*32 + qb
    const int qb = blk & 31;
    const int h  = (blk >> 5) & 7;
    const int b  = blk >> 8;
    const int q0 = qb * 128 + wave * 16;
    const int g = lane >> 4, c = lane & 15;

    // ---- load Q fragment (16 x 64), fold 1/sqrt(DK)=0.125 into it ----
    v16h aq[2];
    {
        const half_t* qrow = Qp + ((size_t)(b * S_) + q0 + c) * D_ + h * DK_;
#pragma unroll
        for (int ch = 0; ch < 2; ch++) {
            V8U lo, hi;
            lo.q = *reinterpret_cast<const uint4*>(qrow + ch * 32 + 8 * g);
            hi.q = *reinterpret_cast<const uint4*>(qrow + ch * 32 + 16 + 8 * g);
            lo.v = lo.v * (half_t)0.125f;   // exact power-of-2 scale in f16
            hi.v = hi.v * (half_t)0.125f;
            V16U a; a.q[0] = lo.q; a.q[1] = hi.q;
            aq[ch] = a.v;
        }
    }

    float m_i[8], l_i[8];
    v8f O[4];
#pragma unroll
    for (int r = 0; r < 8; r++) { m_i[r] = -3.0e38f; l_i[r] = 0.0f; }
#pragma unroll
    for (int t = 0; t < 4; t++) O[t] = v8f{};

    // staging indices: 256 threads cover 32 (j) x 8 (d-chunks of 8 halves)
    const int js = threadIdx.x >> 3;
    const int db = (threadIdx.x & 7) * 8;
    const half_t* kbase = Kp + (size_t)(b * S_) * D_ + h * DK_;
    const half_t* vbase = Vp + (size_t)(b * S_) * D_ + h * DK_;
    half_t* pmine = &Psh[wave * 16 * PP];

    for (int kt = 0; kt < S_ / 32; kt++) {
        const int k0 = kt * 32;
        __syncthreads();
        // ---- stage K tile [32][64]: async direct-to-LDS when available ----
        {
            const half_t* ksrc = kbase + (size_t)(k0 + js) * D_ + db;
#if defined(ASYNC_LDS)
            __builtin_amdgcn_global_load_async_to_lds_b128(
                (AS_GLOBAL gvec4i*)ksrc,
                (AS_LDS gvec4i*)(&Ksh[js * KP + db]),
                0, 0);
#else
            *reinterpret_cast<uint4*>(&Ksh[js * KP + db]) =
                *reinterpret_cast<const uint4*>(ksrc);
#endif
        }
        // ---- stage V tile transposed -> Vsh[d][j] ----
        {
            const half_t* vsrc = vbase + (size_t)(k0 + js) * D_ + db;
#pragma unroll
            for (int i = 0; i < 8; i++) Vsh[(db + i) * VP + js] = vsrc[i];
            // prefetch next V tile (global_prefetch_b8)
            __builtin_prefetch(vbase + (size_t)(k0 + 32 + js) * D_ + db, 0, 1);
        }
        wait_asynccnt0();
        __syncthreads();

        // ---- scores: S(16x32) = Q(16x64) @ K^T(64x32) ----
        v8f sc[2];
#pragma unroll
        for (int tc = 0; tc < 2; tc++) {
            sc[tc] = v8f{};
#pragma unroll
            for (int ch = 0; ch < 2; ch++) {
                V16U bk;
                const half_t* ksrc = &Ksh[(16 * tc + c) * KP + 32 * ch + 8 * g];
                bk.q[0] = *reinterpret_cast<const uint4*>(ksrc);
                bk.q[1] = *reinterpret_cast<const uint4*>(ksrc + 16);
                sc[tc] = wmma32(aq[ch], bk.v, sc[tc]);
            }
        }

        // ---- causal fill (reference: -1e-9, not -inf) ----
#pragma unroll
        for (int tc = 0; tc < 2; tc++) {
            const int j = k0 + 16 * tc + c;
#pragma unroll
            for (int r = 0; r < 8; r++) {
                const int sq = q0 + r + 8 * g;
                if (j > sq) sc[tc][r] = -1e-9f;
            }
        }

        // ---- row max (16 lanes of this half hold the 16 cols of row r+8g) --
        float mx[8];
#pragma unroll
        for (int r = 0; r < 8; r++) mx[r] = fmaxf(sc[0][r], sc[1][r]);
#pragma unroll
        for (int off = 8; off >= 1; off >>= 1)
#pragma unroll
            for (int r = 0; r < 8; r++) mx[r] = fmaxf(mx[r], __shfl_xor(mx[r], off, 32));

        float alpha[8], rs[8];
#pragma unroll
        for (int r = 0; r < 8; r++) {
            const float mn = fmaxf(m_i[r], mx[r]);
            alpha[r] = __expf(m_i[r] - mn);
            m_i[r] = mn;
            rs[r] = 0.0f;
        }

        // ---- P = exp(S - m), store to per-wave LDS for layout change ----
#pragma unroll
        for (int tc = 0; tc < 2; tc++)
#pragma unroll
            for (int r = 0; r < 8; r++) {
                const float p = __expf(sc[tc][r] - m_i[r]);
                rs[r] += p;
                pmine[(r + 8 * g) * PP + 16 * tc + c] = (half_t)p;
            }
#pragma unroll
        for (int off = 8; off >= 1; off >>= 1)
#pragma unroll
            for (int r = 0; r < 8; r++) rs[r] += __shfl_xor(rs[r], off, 32);
#pragma unroll
        for (int r = 0; r < 8; r++) l_i[r] = l_i[r] * alpha[r] + rs[r];
#pragma unroll
        for (int t = 0; t < 4; t++)
#pragma unroll
            for (int r = 0; r < 8; r++) O[t][r] *= alpha[r];

        // ---- reload P as A-fragment (16x32) ----
        v16h ap;
        {
            const half_t* psrc = &pmine[c * PP];
            V16U a;
            a.q[0] = *reinterpret_cast<const uint4*>(psrc + 8 * g);
            a.q[1] = *reinterpret_cast<const uint4*>(psrc + 16 + 8 * g);
            ap = a.v;
        }
        // ---- O(16x64) += P(16x32) @ V(32x64) ----
#pragma unroll
        for (int t = 0; t < 4; t++) {
            V16U bv;
            const half_t* vs = &Vsh[(16 * t + c) * VP + 8 * g];
            bv.q[0] = *reinterpret_cast<const uint4*>(vs);
            bv.q[1] = *reinterpret_cast<const uint4*>(vs + 16);
            O[t] = wmma32(ap, bv.v, O[t]);
        }
    }

    // ---- epilogue: normalize, store f16 [B,S,D] ----
#pragma unroll
    for (int t = 0; t < 4; t++)
#pragma unroll
        for (int r = 0; r < 8; r++)
            Xo[((size_t)(b * S_) + q0 + r + 8 * g) * D_ + h * DK_ + 16 * t + c] =
                (half_t)(O[t][r] / l_i[r]);
}

// ---------------------------------------------------------------------------
// Inputs: 0 q, 1 k, 2 v, 3 mask(unused, causal computed analytically),
//         4 wq, 5 bq, 6 wk, 7 bk, 8 wv, 9 bv, 10 wo, 11 bo
// ---------------------------------------------------------------------------
extern "C" void kernel_launch(void* const* d_in, const int* in_sizes, int n_in,
                              void* d_out, int out_size, void* d_ws, size_t ws_size,
                              hipStream_t stream) {
    const float* q  = (const float*)d_in[0];
    const float* k  = (const float*)d_in[1];
    const float* v  = (const float*)d_in[2];
    const float* wq = (const float*)d_in[4];
    const float* bq = (const float*)d_in[5];
    const float* wk = (const float*)d_in[6];
    const float* bk = (const float*)d_in[7];
    const float* wv = (const float*)d_in[8];
    const float* bv = (const float*)d_in[9];
    const float* wo = (const float*)d_in[10];
    const float* bo = (const float*)d_in[11];
    float* out = (float*)d_out;

    const size_t N = (size_t)M_ * D_;     // 4,194,304 elements
    half_t* qh   = (half_t*)d_ws;
    half_t* kh   = qh + N;
    half_t* vh   = kh + N;
    half_t* xatt = vh + N;

    const int proj_blocks = (M_ / 64) * (D_ / 64) / 8;   // 128 blocks, 1024 waves
    proj_f32h<<<proj_blocks, 256, 0, stream>>>(q, wq, bq, qh);
    proj_f32h<<<proj_blocks, 256, 0, stream>>>(k, wk, bk, kh);
    proj_f32h<<<proj_blocks, 256, 0, stream>>>(v, wv, bv, vh);

    attn_kernel<<<B_ * H_ * (S_ / 128), 256, 0, stream>>>(qh, kh, vh, xatt);

    proj_h32<<<proj_blocks, 256, 0, stream>>>(xatt, wo, bo, out);
}
